// GPTCore_15814069584060
// MI455X (gfx1250) — matrix operand order repaired and checked by reading, not compiled, not run on hardware
//
#include <hip/hip_runtime.h>

// ---------------------------------------------------------------------------
// Types
// ---------------------------------------------------------------------------
typedef __bf16 bf16;
typedef __attribute__((ext_vector_type(16))) __bf16 v16bf;
typedef __attribute__((ext_vector_type(8)))  __bf16 v8bf;
typedef __attribute__((ext_vector_type(8)))  float  v8f;
typedef __attribute__((ext_vector_type(4)))  float  v4f;
typedef __attribute__((ext_vector_type(4)))  unsigned u32x4;
typedef __attribute__((ext_vector_type(8)))  int      i32x8;
typedef __attribute__((ext_vector_type(4)))  int      i32x4;

#define S_LEN 2048
#define DMODEL 768
#define NHEAD 6
#define HDIM 128
#define NLAYER 8
#define VOCAB 50257
#define EOS_TOK 50256
#define EPS_RMS 1.1920929e-07f
#define ATTN_SCALE 0.12f
// |score| = |0.12 * <rms(q)_rope, rms(k)_rope>| <= 0.12*128 = 15.36 < 16,
// so softmax with a fixed max of 16 is exact (no online max needed).
#define SMAX 16.0f

#if __has_builtin(__builtin_amdgcn_tensor_load_to_lds) && \
    __has_builtin(__builtin_amdgcn_s_wait_tensorcnt)
#define HAVE_TDM 1
#else
#define HAVE_TDM 0
#endif

// ---------------------------------------------------------------------------
// WMMA helpers
// ---------------------------------------------------------------------------
__device__ __forceinline__ v8f wmma_bf16(v16bf a, v16bf b, v8f c) {
  return __builtin_amdgcn_wmma_f32_16x16x32_bf16(false, a, false, b, (short)0, c,
                                                 false, false);
}

// A fragment (16x32, M x K), source row-major [16, ld] bf16, K contiguous.
// lane l: row = l&15, K = (l>>4)*16 + {0..15} contiguous.
__device__ __forceinline__ v16bf load_afrag(const bf16* p, int ld) {
  int l = threadIdx.x & 31;
  const bf16* q = p + (size_t)(l & 15) * ld + ((l >> 4) << 4);
  v8bf a0 = *(const v8bf*)q;
  v8bf a1 = *(const v8bf*)(q + 8);
  v16bf r;
#pragma unroll
  for (int j = 0; j < 8; ++j) { r[j] = a0[j]; r[j + 8] = a1[j]; }
  return r;
}

// B fragment (32x16, K x N) from W row-major [N, ld] bf16 (C = A*W^T).
// lane l: column n = l&15 (W row n), K = (l>>4)*16 + {0..15} contiguous.
__device__ __forceinline__ v16bf load_bfrag(const bf16* __restrict__ p, int ld) {
  int l = threadIdx.x & 31;
  const bf16* q = p + (size_t)(l & 15) * ld + ((l >> 4) << 4);
  v8bf b0 = *(const v8bf*)q;
  v8bf b1 = *(const v8bf*)(q + 8);
  v16bf r;
#pragma unroll
  for (int j = 0; j < 8; ++j) { r[j] = b0[j]; r[j + 8] = b1[j]; }
  return r;
}

// bf16 B fragment with row clamp (for ragged N)
__device__ __forceinline__ v16bf load_bfrag_clamp(const bf16* __restrict__ W,
                                                  int n0, int kk, int ld, int N) {
  int l = threadIdx.x & 31;
  int n = n0 + (l & 15);
  if (n >= N) n = N - 1;
  const bf16* q = W + (size_t)n * ld + kk + ((l >> 4) << 4);
  v8bf b0 = *(const v8bf*)q;
  v8bf b1 = *(const v8bf*)(q + 8);
  v16bf r;
#pragma unroll
  for (int j = 0; j < 8; ++j) { r[j] = b0[j]; r[j + 8] = b1[j]; }
  return r;
}

// f32 B fragment converted inline to bf16 (weights stay f32; L2-resident)
__device__ __forceinline__ v16bf load_bfrag_f32(const float* __restrict__ W,
                                                int n0, int kk, int ld) {
  int l = threadIdx.x & 31;
  int n = n0 + (l & 15);
  const float* q = W + (size_t)n * ld + kk + ((l >> 4) << 4);
  v16bf r;
#pragma unroll
  for (int j = 0; j < 4; ++j) {
    v4f f = *(const v4f*)(q + 4 * j);
    r[4 * j + 0] = (bf16)f[0];
    r[4 * j + 1] = (bf16)f[1];
    r[4 * j + 2] = (bf16)f[2];
    r[4 * j + 3] = (bf16)f[3];
  }
  return r;
}

#if HAVE_TDM
// Issue one TDM 2D tile load: tile_dim = [32 (K), 128 (M)] bf16 elements
// from a row-major [M, K_] bf16 tensor into LDS (row-major 128x32).
__device__ __forceinline__ void tdm_load_a_tile(const void* lds_dst,
                                                const void* gsrc, int K_) {
  unsigned lds_off = (unsigned)(size_t)lds_dst;   // low 32 bits = LDS offset
  unsigned long long ga = (unsigned long long)(size_t)gsrc;
  u32x4 g0;
  g0[0] = 1u;                                  // count=1, user mode, no gather
  g0[1] = lds_off;                             // lds_addr
  g0[2] = (unsigned)ga;                        // global_addr[31:0]
  g0[3] = (unsigned)((ga >> 32) & 0x1FFFFFFu)  // global_addr[56:32]
          | (2u << 30);                        // type = 2 (image)
  i32x8 g1;
  g1[0] = 0x00010000;                          // data_size = 1 (2 bytes)
  g1[1] = (int)((K_ & 0xFFFF) << 16);          // tensor_dim0[15:0]
  g1[2] = (int)(((unsigned)K_ >> 16) & 0xFFFFu) |
          (int)((S_LEN & 0xFFFF) << 16);       // dim0[31:16] | dim1[15:0]
  g1[3] = (int)(32u << 16);                    // dim1[31:16]=0 | tile_dim0=32
  g1[4] = 128;                                 // tile_dim1=128, tile_dim2=0
  g1[5] = K_;                                  // tensor_dim0_stride[31:0]
  g1[6] = 0;                                   // stride0[47:32] | stride1 lo
  g1[7] = 0;
  i32x4 g2 = {};
  i32x4 g3 = {};
#if defined(__clang_major__) && __clang_major__ >= 23
  i32x8 g4 = {};
  __builtin_amdgcn_tensor_load_to_lds(g0, g1, g2, g3, g4, 0);
#else
  __builtin_amdgcn_tensor_load_to_lds(g0, g1, g2, g3, 0);
#endif
}
#endif

// ---------------------------------------------------------------------------
// Reductions
// ---------------------------------------------------------------------------
__device__ __forceinline__ float wave_sum32(float v) {
#pragma unroll
  for (int m = 1; m < 32; m <<= 1) v += __shfl_xor(v, m, 32);
  return v;
}
__device__ __forceinline__ float half_sum16(float v) {
#pragma unroll
  for (int m = 1; m < 16; m <<= 1) v += __shfl_xor(v, m, 32);
  return v;
}
__device__ __forceinline__ float block_sum256(float v) {
  __shared__ float red[8];
  v = wave_sum32(v);
  int w = threadIdx.x >> 5, lane = threadIdx.x & 31;
  if (lane == 0) red[w] = v;
  __syncthreads();
  float t = 0.f;
#pragma unroll
  for (int j = 0; j < 8; ++j) t += red[j];
  __syncthreads();
  return t;
}

// ---------------------------------------------------------------------------
// f32 -> bf16 bulk conversion (grid-stride, 8 elems/thread/iter)
// ---------------------------------------------------------------------------
__global__ __launch_bounds__(256) void k_cvt_bf16(const float* __restrict__ src,
                                                  bf16* __restrict__ dst,
                                                  long n) {
  long stride = (long)gridDim.x * 256 * 8;
  for (long i = ((long)blockIdx.x * 256 + threadIdx.x) * 8; i < n; i += stride) {
    v4f f0 = *(const v4f*)(src + i);
    v4f f1 = *(const v4f*)(src + i + 4);
    v8bf o;
#pragma unroll
    for (int j = 0; j < 4; ++j) { o[j] = (bf16)f0[j]; o[j + 4] = (bf16)f1[j]; }
    *(v8bf*)(dst + i) = o;
  }
}

// ---------------------------------------------------------------------------
// docs cumsum + block masks (both windows), single thread (tiny work)
// ---------------------------------------------------------------------------
__global__ void k_docs_mask(const int* __restrict__ in_seq,
                            const int* __restrict__ swnb,
                            int* __restrict__ docs,
                            unsigned* __restrict__ tf,
                            unsigned* __restrict__ tp) {
  int c = 0;
  for (int s = 0; s < S_LEN; ++s) {
    c += (in_seq[s] == EOS_TOK) ? 1 : 0;
    docs[s] = c;
  }
  int dl[16], dh[16];
  for (int b = 0; b < 16; ++b) {
    dl[b] = docs[b * 128];
    dh[b] = docs[b * 128 + 127];
  }
  for (int win = 0; win < 2; ++win) {
    int w = (win == 0) ? swnb[0] : (swnb[0] / 2);
    for (int i = 0; i < 16; ++i) {
      unsigned many = 0, mall = 0;
      for (int j = 0; j < 16; ++j) {
        bool any = (i >= j) && (dl[i] <= dh[j]) && (dh[i] >= dl[j]);
        bool all = (i > j) && (dl[i] == dh[j]) && (dh[i] == dl[j]);
        if (any) many |= 1u << j;
        if (all) mall |= 1u << j;
      }
      unsigned part = many & ~mall;
      int nf = __popc(mall);
      int kcf = min(nf, w - 1);
      int kcp = min(__popc(part), max(w - nf, 1));
      unsigned kf = 0, kp = 0;
      int cf = 0, cp = 0;
      for (int j = 15; j >= 0; --j) {
        if ((mall >> j) & 1) { cf++; if (cf <= kcf) kf |= 1u << j; }
        if ((part >> j) & 1) { cp++; if (cp <= kcp) kp |= 1u << j; }
      }
      tf[win * 16 + i] = kf;
      tp[win * 16 + i] = kp;
    }
  }
}

// ---------------------------------------------------------------------------
// x0 = x = rms(embed[input_seq])
// ---------------------------------------------------------------------------
__global__ __launch_bounds__(256) void k_embed(const int* __restrict__ in_seq,
                                               const float* __restrict__ embed,
                                               float* __restrict__ x,
                                               float* __restrict__ x0) {
  int s = blockIdx.x;
  int tok = in_seq[s];
  const float* row = embed + (size_t)tok * DMODEL;
  float vals[3], ss = 0.f;
#pragma unroll
  for (int u = 0; u < 3; ++u) {
    int d = threadIdx.x + 256 * u;
    vals[u] = row[d];
    ss += vals[u] * vals[u];
  }
  float tot = block_sum256(ss);
  float r = rsqrtf(tot * (1.0f / DMODEL) + EPS_RMS);
#pragma unroll
  for (int u = 0; u < 3; ++u) {
    int d = threadIdx.x + 256 * u;
    float v = vals[u] * r;
    x[(size_t)s * DMODEL + d] = v;
    x0[(size_t)s * DMODEL + d] = v;
  }
}

// ---------------------------------------------------------------------------
// x = lam0*(x + sw*skip) + lam1*x0 ; xab = bf16(rms(x))
// ---------------------------------------------------------------------------
__global__ __launch_bounds__(256) void k_prelayer(float* __restrict__ x,
                                                  const float* __restrict__ x0,
                                                  const float* skip,
                                                  const float* __restrict__ scalars,
                                                  int layer, int skipidx,
                                                  bf16* __restrict__ xab) {
  int s = blockIdx.x;
  float lam0 = scalars[NLAYER + 2 * layer];
  float lam1 = scalars[NLAYER + 2 * layer + 1];
  float sw = skip ? scalars[skipidx] : 0.f;
  float vals[3], ss = 0.f;
#pragma unroll
  for (int u = 0; u < 3; ++u) {
    int d = threadIdx.x + 256 * u;
    size_t idx = (size_t)s * DMODEL + d;
    float xv = x[idx];
    if (skip) xv += sw * skip[idx];
    xv = lam0 * xv + lam1 * x0[idx];
    vals[u] = xv;
    ss += xv * xv;
  }
  float tot = block_sum256(ss);
  float r = rsqrtf(tot * (1.0f / DMODEL) + EPS_RMS);
#pragma unroll
  for (int u = 0; u < 3; ++u) {
    int d = threadIdx.x + 256 * u;
    size_t idx = (size_t)s * DMODEL + d;
    x[idx] = vals[u];
    xab[idx] = (bf16)(vals[u] * r);
  }
}

// xab = bf16(rms(x)), x unchanged
__global__ __launch_bounds__(256) void k_rms_bf(const float* __restrict__ x,
                                                bf16* __restrict__ xab) {
  int s = blockIdx.x;
  float vals[3], ss = 0.f;
#pragma unroll
  for (int u = 0; u < 3; ++u) {
    int d = threadIdx.x + 256 * u;
    vals[u] = x[(size_t)s * DMODEL + d];
    ss += vals[u] * vals[u];
  }
  float tot = block_sum256(ss);
  float r = rsqrtf(tot * (1.0f / DMODEL) + EPS_RMS);
#pragma unroll
  for (int u = 0; u < 3; ++u) {
    int d = threadIdx.x + 256 * u;
    xab[(size_t)s * DMODEL + d] = (bf16)(vals[u] * r);
  }
}

// ---------------------------------------------------------------------------
// QKV post: per-head rms+rope for q,k (q pre-scaled by ATTN_SCALE);
// v = sal0*v + sal1*ve, written transposed as VT[h][d][s]
// ---------------------------------------------------------------------------
__global__ __launch_bounds__(256) void k_qkvpost(const float* __restrict__ qkv,
                                                 const int* __restrict__ in_seq,
                                                 const float* __restrict__ ve_embeds,
                                                 const float* __restrict__ scalars,
                                                 int layer, int vei,
                                                 bf16* __restrict__ Qb,
                                                 bf16* __restrict__ Kb,
                                                 bf16* __restrict__ VT) {
  int s = blockIdx.x;
  int lane = threadIdx.x & 31, w = threadIdx.x >> 5;
  const float* row = qkv + (size_t)s * (3 * DMODEL);
  float sal0 = scalars[3 * NLAYER + 2 * layer];
  float sal1 = scalars[3 * NLAYER + 2 * layer + 1];

  for (int t = w; t < 18; t += 8) {
    int kind = t / 6, h = t % 6;
    if (kind < 2) {
      const float* src = row + (kind * 6 + h) * HDIM;
      float x1a = src[lane], x1b = src[lane + 32];
      float x2a = src[lane + 64], x2b = src[lane + 96];
      float ss = x1a * x1a + x1b * x1b + x2a * x2a + x2b * x2b;
      ss = wave_sum32(ss);
      float r = rsqrtf(ss * (1.0f / HDIM) + EPS_RMS);
      x1a *= r; x1b *= r; x2a *= r; x2b *= r;
      // rope: pair (j, j+64); freq nonzero only for j<32 (== lane)
      float f = powf(1.0f / 1024.0f, (float)lane * (1.0f / 31.0f));
      float th = (float)s * f;
      float cs = __cosf(th), sn = __sinf(th);
      float o1a = x1a * cs + x2a * sn;
      float o2a = -x1a * sn + x2a * cs;
      // j = lane+32 has freq 0 -> identity
      float scale = (kind == 0) ? ATTN_SCALE : 1.0f;
      bf16* dst = ((kind == 0) ? Qb : Kb) + (size_t)s * DMODEL + h * HDIM;
      dst[lane]      = (bf16)(o1a * scale);
      dst[lane + 32] = (bf16)(x1b * scale);
      dst[lane + 64] = (bf16)(o2a * scale);
      dst[lane + 96] = (bf16)(x2b * scale);
    } else {
      const float* src = row + (12 + h) * HDIM;
      int tok = in_seq[s];
      const float* vp =
          (vei >= 0) ? ve_embeds + ((size_t)vei * VOCAB + tok) * DMODEL + h * HDIM
                     : nullptr;
#pragma unroll
      for (int jj = 0; jj < 4; ++jj) {
        int j = lane + 32 * jj;
        float val = sal0 * src[j];
        if (vp) val += sal1 * vp[j];
        VT[(size_t)(h * HDIM + j) * S_LEN + s] = (bf16)val;
      }
    }
  }
}

// ---------------------------------------------------------------------------
// Flash attention, fixed-max softmax (scores provably < 16).
// One wave per 16 q rows, per head. Block mask + token mask.
// ---------------------------------------------------------------------------
__global__ __launch_bounds__(256) void k_attn(const bf16* __restrict__ Q,
                                              const bf16* __restrict__ Kb,
                                              const bf16* __restrict__ VT,
                                              const int* __restrict__ docs,
                                              const unsigned* __restrict__ tfb,
                                              const unsigned* __restrict__ tpb,
                                              bf16* __restrict__ Y) {
  __shared__ bf16 pls[8][16 * 48];
  int lane = threadIdx.x & 31, w = threadIdx.x >> 5;
  int half = lane >> 4, ln = lane & 15;
  int head = blockIdx.y;
  int qb = blockIdx.x;  // 128-row block index (== mask block)
  int qbase = qb * 128 + w * 16;

  v16bf aq[4];
#pragma unroll
  for (int t = 0; t < 4; ++t)
    aq[t] = load_afrag(Q + (size_t)qbase * DMODEL + head * HDIM + t * 32, DMODEL);

  int qidx[8], dq[8];
#pragma unroll
  for (int v = 0; v < 8; ++v) {
    qidx[v] = qbase + 8 * half + v;
    dq[v] = docs[qidx[v]];
  }

  float ls[8] = {0.f, 0.f, 0.f, 0.f, 0.f, 0.f, 0.f, 0.f};
  v8f acc[8] = {};

  unsigned fbits = tfb[qb], pbits = tpb[qb];
  bf16* pw = pls[w];

  for (int kb = 0; kb < 16; ++kb) {
    bool full = (fbits >> kb) & 1;
    bool part = (pbits >> kb) & 1;
    if (!full && !part) continue;
    for (int ko = 0; ko < 4; ++ko) {
      int kbase = kb * 128 + ko * 32;
      v8f s0 = {}, s1 = {};
#pragma unroll
      for (int t = 0; t < 4; ++t) {
        v16bf b0 = load_bfrag(Kb + (size_t)kbase * DMODEL + head * HDIM + t * 32, DMODEL);
        v16bf b1 = load_bfrag(Kb + (size_t)(kbase + 16) * DMODEL + head * HDIM + t * 32, DMODEL);
        s0 = wmma_bf16(aq[t], b0, s0);
        s1 = wmma_bf16(aq[t], b1, s1);
      }
      float p0[8], p1[8];
      if (full) {
#pragma unroll
        for (int v = 0; v < 8; ++v) {
          p0[v] = __expf(s0[v] - SMAX);
          p1[v] = __expf(s1[v] - SMAX);
        }
      } else {
        int k0 = kbase + ln, k1 = k0 + 16;
        int dk0 = docs[k0], dk1 = docs[k1];
#pragma unroll
        for (int v = 0; v < 8; ++v) {
          p0[v] = ((qidx[v] >= k0) && (dq[v] == dk0)) ? __expf(s0[v] - SMAX) : 0.f;
          p1[v] = ((qidx[v] >= k1) && (dq[v] == dk1)) ? __expf(s1[v] - SMAX) : 0.f;
        }
      }
#pragma unroll
      for (int v = 0; v < 8; ++v) ls[v] += p0[v] + p1[v];
      // stage P through LDS (D-layout -> A-layout)
#pragma unroll
      for (int v = 0; v < 8; ++v) {
        pw[(8 * half + v) * 48 + ln] = (bf16)p0[v];
        pw[(8 * half + v) * 48 + 16 + ln] = (bf16)p1[v];
      }
      asm volatile("s_wait_dscnt 0" ::: "memory");
      const bf16* qp = pw + ln * 48 + 16 * half;
      v16bf pf;
#pragma unroll
      for (int j = 0; j < 8; ++j) { pf[j] = qp[j]; pf[j + 8] = qp[j + 8]; }
#pragma unroll
      for (int t = 0; t < 8; ++t) {
        v16bf bv = load_bfrag(VT + (size_t)(head * HDIM + t * 16) * S_LEN + kbase, S_LEN);
        acc[t] = wmma_bf16(pf, bv, acc[t]);
      }
    }
  }
  // finalize row sums (per-lane partials -> full rows), once
  float inv[8];
#pragma unroll
  for (int v = 0; v < 8; ++v) {
    float t = half_sum16(ls[v]);
    inv[v] = (t > 0.f) ? (1.0f / t) : 0.f;
  }
#pragma unroll
  for (int t = 0; t < 8; ++t)
#pragma unroll
    for (int v = 0; v < 8; ++v) {
      int r = qbase + 8 * half + v;
      int c = head * HDIM + t * 16 + ln;
      Y[(size_t)r * DMODEL + c] = (bf16)(acc[t][v] * inv[v]);
    }
}

// ---------------------------------------------------------------------------
// Generic NT GEMM: C[M,N] = A[M,K](bf16) * W[N,K]^T.  M == 2048 always.
// A tile (128x32 bf16) is TDM-staged into LDS double buffers: wave 0 issues
// tensor_load_to_lds for tile it+1 while all 8 waves consume tile it.
// W is f32 (converted inline) except EPI 3 which uses a bf16 W (Wb).
// Epilogues: 0=store f32, 1=residual add (+optional copy), 2=relu^2->bf16,
//            3=fused softcap cross-entropy statistics
// ---------------------------------------------------------------------------
template <int EPI>
__global__ __launch_bounds__(256) void k_gemm(const bf16* __restrict__ A,
                                              const float* __restrict__ W,
                                              const bf16* __restrict__ Wb,
                                              int M, int N, int K,
                                              float* outf, const float* resid,
                                              float* copyp, bf16* outb,
                                              float* sumexp, float* capt,
                                              const int* target) {
  int lane = threadIdx.x & 31, w = threadIdx.x >> 5;
  int half = lane >> 4, ln = lane & 15;
  int row0 = blockIdx.y * 128 + (w >> 1) * 32;
  int col0 = blockIdx.x * 64 + (w & 1) * 32;
  v8f acc[2][2] = {};
  int nk = K / 32;

#if HAVE_TDM
  __shared__ bf16 atile[2][128 * 32];
  const bf16* Ablk = A + (size_t)(blockIdx.y * 128) * K;
  if (w == 0) tdm_load_a_tile(&atile[0][0], Ablk, K);
#endif

  for (int it = 0; it < nk; ++it) {
    int kk = it * 32;
#if HAVE_TDM
    if (w == 0) {
      if (it + 1 < nk) {
        tdm_load_a_tile(&atile[(it + 1) & 1][0], Ablk + (size_t)(kk + 32), K);
        __builtin_amdgcn_s_wait_tensorcnt(1);
      } else {
        __builtin_amdgcn_s_wait_tensorcnt(0);
      }
    }
    __syncthreads();
    const bf16* Asrc = &atile[it & 1][(w >> 1) * 32 * 32];
    v16bf a0 = load_afrag(Asrc, 32);
    v16bf a1 = load_afrag(Asrc + 16 * 32, 32);
#else
    v16bf a0 = load_afrag(A + (size_t)row0 * K + kk, K);
    v16bf a1 = load_afrag(A + (size_t)(row0 + 16) * K + kk, K);
#endif
    v16bf b0, b1;
    if constexpr (EPI == 3) {
      b0 = load_bfrag_clamp(Wb, col0, kk, K, N);
      b1 = load_bfrag_clamp(Wb, col0 + 16, kk, K, N);
    } else {
      if (kk + 32 < K) __builtin_prefetch(W + (size_t)col0 * K + kk + 32, 0, 3);
      b0 = load_bfrag_f32(W, col0, kk, K);
      b1 = load_bfrag_f32(W, col0 + 16, kk, K);
    }
    acc[0][0] = wmma_bf16(a0, b0, acc[0][0]);
    acc[0][1] = wmma_bf16(a0, b1, acc[0][1]);
    acc[1][0] = wmma_bf16(a1, b0, acc[1][0]);
    acc[1][1] = wmma_bf16(a1, b1, acc[1][1]);
#if HAVE_TDM
    __syncthreads();  // all waves done reading atile[it&1] before reuse
#endif
  }

  if constexpr (EPI == 3) {
    // softcap cross-entropy statistics; exp(cap-15) row sums + target capture
#pragma unroll
    for (int tr = 0; tr < 2; ++tr) {
      int rbase = row0 + tr * 16 + 8 * half;
#pragma unroll
      for (int v = 0; v < 8; ++v) {
        int r = rbase + v;
        int tgt = target[r];
        float e = 0.f;
#pragma unroll
        for (int tc = 0; tc < 2; ++tc) {
          int c = col0 + tc * 16 + ln;
          if (c < N) {
            float z = acc[tr][tc][v];
            float cap = 15.f * z * rsqrtf(z * z + 225.f);
            e += __expf(cap - 15.f);
            if (c == tgt) atomicAdd(&capt[r], cap);
          }
        }
        e = half_sum16(e);
        if (ln == v) atomicAdd(&sumexp[r], e);
      }
    }
  } else {
#pragma unroll
    for (int tr = 0; tr < 2; ++tr)
#pragma unroll
      for (int tc = 0; tc < 2; ++tc)
#pragma unroll
        for (int v = 0; v < 8; ++v) {
          int r = row0 + tr * 16 + 8 * half + v;
          int c = col0 + tc * 16 + ln;
          size_t idx = (size_t)r * N + c;
          float z = acc[tr][tc][v];
          if constexpr (EPI == 0) {
            outf[idx] = z;
          } else if constexpr (EPI == 1) {
            float val = resid[idx] + z;
            outf[idx] = val;
            if (copyp) copyp[idx] = val;
          } else if constexpr (EPI == 2) {
            float rl = fmaxf(z, 0.f);
            outb[idx] = (bf16)(rl * rl);
          }
        }
  }
}

// ---------------------------------------------------------------------------
// final loss reduction
// ---------------------------------------------------------------------------
__global__ __launch_bounds__(256) void k_loss(const float* __restrict__ sumexp,
                                              const float* __restrict__ capt,
                                              float* __restrict__ out) {
  float acc = 0.f;
  for (int s = threadIdx.x; s < S_LEN; s += 256)
    acc += capt[s] - 15.f - logf(sumexp[s]);
  float tot = block_sum256(acc);
  if (threadIdx.x == 0) out[0] = -tot * (1.0f / S_LEN);
}

// ---------------------------------------------------------------------------
// host launcher
// ---------------------------------------------------------------------------
extern "C" void kernel_launch(void* const* d_in, const int* in_sizes, int n_in,
                              void* d_out, int out_size, void* d_ws, size_t ws_size,
                              hipStream_t stream) {
  const int* input_seq = (const int*)d_in[0];
  const int* swnb = (const int*)d_in[1];
  const int* target = (const int*)d_in[2];
  const float* embed = (const float*)d_in[3];
  const float* ve_embeds = (const float*)d_in[4];
  const float* qkv_w = (const float*)d_in[5];
  const float* aproj_w = (const float*)d_in[6];
  const float* fc_w = (const float*)d_in[7];
  const float* mproj_w = (const float*)d_in[8];
  const float* scalars = (const float*)d_in[9];

  char* p = (char*)d_ws;
  auto alloc = [&](size_t bytes) {
    void* r = (void*)p;
    p += (bytes + 255) & ~(size_t)255;
    return r;
  };
  int* docs = (int*)alloc(S_LEN * 4);
  unsigned* tf = (unsigned*)alloc(2 * 16 * 4);
  unsigned* tp = (unsigned*)alloc(2 * 16 * 4);
  float* x = (float*)alloc((size_t)S_LEN * DMODEL * 4);
  float* x0 = (float*)alloc((size_t)S_LEN * DMODEL * 4);
  float* skipb[3];
  for (int j = 0; j < 3; ++j) skipb[j] = (float*)alloc((size_t)S_LEN * DMODEL * 4);
  bf16* xab = (bf16*)alloc((size_t)S_LEN * DMODEL * 2);
  float* qkvf = (float*)alloc((size_t)S_LEN * 3 * DMODEL * 4);
  bf16* qb = (bf16*)alloc((size_t)S_LEN * DMODEL * 2);
  bf16* kb = (bf16*)alloc((size_t)S_LEN * DMODEL * 2);
  bf16* vt = (bf16*)alloc((size_t)S_LEN * DMODEL * 2);
  bf16* yb = (bf16*)alloc((size_t)S_LEN * DMODEL * 2);
  bf16* hb = (bf16*)alloc((size_t)S_LEN * 4 * DMODEL * 2);
  float* sumexp = (float*)alloc(S_LEN * 4);
  float* capt = (float*)alloc(S_LEN * 4);
  bf16* embed_bf = (bf16*)alloc((size_t)VOCAB * DMODEL * 2);

  k_docs_mask<<<1, 1, 0, stream>>>(input_seq, swnb, docs, tf, tp);
  k_embed<<<S_LEN, 256, 0, stream>>>(input_seq, embed, x, x0);
  // embed -> bf16 once per call (matches reference's bf16 cast for logits,
  // and keeps the vocab sweep L2-resident: 77MB bf16 vs 154MB f32)
  k_cvt_bf16<<<4096, 256, 0, stream>>>(embed, embed_bf,
                                       (long)VOCAB * DMODEL);

  for (int i = 0; i < NLAYER; ++i) {
    const float* skipsrc = nullptr;
    int sidx = 0;
    if (i == 5) { skipsrc = skipb[0]; sidx = 1; }
    if (i == 6) { skipsrc = skipb[1]; sidx = 2; }
    if (i == 7) { skipsrc = skipb[2]; sidx = 3; }
    k_prelayer<<<S_LEN, 256, 0, stream>>>(x, x0, skipsrc, scalars, i, sidx, xab);

    // QKV: [2048,2304] = xab @ qkv_w[i]^T
    k_gemm<0><<<dim3(36, 16), 256, 0, stream>>>(
        xab, qkv_w + (size_t)i * 3 * DMODEL * DMODEL, nullptr, S_LEN, 3 * DMODEL,
        DMODEL, qkvf, nullptr, nullptr, nullptr, nullptr, nullptr, nullptr);

    int vei = (i < 3) ? i : ((i >= 5) ? i - 5 : -1);
    k_qkvpost<<<S_LEN, 256, 0, stream>>>(qkvf, input_seq, ve_embeds, scalars, i,
                                         vei, qb, kb, vt);

    bool lng = (i == 0 || i == 4 || i == 7);
    k_attn<<<dim3(16, NHEAD), 256, 0, stream>>>(
        qb, kb, vt, docs, tf + (lng ? 0 : 16), tp + (lng ? 0 : 16), yb);

    // x += y @ attn_proj^T
    k_gemm<1><<<dim3(12, 16), 256, 0, stream>>>(
        yb, aproj_w + (size_t)i * DMODEL * DMODEL, nullptr, S_LEN, DMODEL, DMODEL,
        x, x, nullptr, nullptr, nullptr, nullptr, nullptr);

    k_rms_bf<<<S_LEN, 256, 0, stream>>>(x, xab);

    // h = relu(xab @ fc^T)^2  -> bf16
    k_gemm<2><<<dim3(48, 16), 256, 0, stream>>>(
        xab, fc_w + (size_t)i * 4 * DMODEL * DMODEL, nullptr, S_LEN, 4 * DMODEL,
        DMODEL, nullptr, nullptr, nullptr, hb, nullptr, nullptr, nullptr);

    // x += h @ mlp_proj^T ; record skip for layers 1..3
    float* cpy = (i == 1) ? skipb[0] : (i == 2) ? skipb[1] : (i == 3) ? skipb[2] : nullptr;
    k_gemm<1><<<dim3(12, 16), 256, 0, stream>>>(
        hb, mproj_w + (size_t)i * DMODEL * 4 * DMODEL, nullptr, S_LEN, DMODEL,
        4 * DMODEL, x, x, cpy, nullptr, nullptr, nullptr, nullptr);
  }

  k_rms_bf<<<S_LEN, 256, 0, stream>>>(x, xab);
  hipMemsetAsync(sumexp, 0, S_LEN * 4, stream);
  hipMemsetAsync(capt, 0, S_LEN * 4, stream);

  // logits + softcap + CE statistics, streamed over vocab (N ragged)
  k_gemm<3><<<dim3((VOCAB + 63) / 64, 16), 256, 0, stream>>>(
      xab, nullptr, embed_bf, S_LEN, VOCAB, DMODEL, nullptr, nullptr, nullptr,
      nullptr, sumexp, capt, target);

  k_loss<<<1, 256, 0, stream>>>(sumexp, capt, (float*)d_out);
}
